// MoE_56375740727790
// MI455X (gfx1250) — matrix operand order repaired and checked by reading, not compiled, not run on hardware
//
#include <hip/hip_runtime.h>
#include <math.h>

// ---------------------------------------------------------------------------
// MoE FFN (top-2 of 8 experts), T=8192 tokens, D=1024, H=4096.
// bf16 WMMA (v_wmma_f32_16x16x32_bf16) GEMMs, f32 accumulation.
// Block tile 128(M) x 256(N), K-step 64; 8 waves, each wave owns a 64x64
// register tile = 4x4 WMMA accumulators (16 wmma / 32-deep K chunk).
// A tiles staged global->LDS with CDNA5 async LDS loads
// (global_load_async_to_lds_b128 + s_wait_asynccnt).
// ---------------------------------------------------------------------------

#define T_TOK   8192
#define DIM_D   1024
#define DIM_H   4096
#define N_EXP   8

typedef __attribute__((ext_vector_type(16))) __bf16        v16bf;
typedef __attribute__((ext_vector_type(8)))  float         v8f;
typedef __attribute__((ext_vector_type(4)))  unsigned int  u32x4;

union Frag  { u32x4 q[2]; v16bf v; };
union PackU { unsigned short s[8]; u32x4 q; };

typedef __attribute__((address_space(3))) unsigned short lds_us;

// ---- workspace byte offsets ----
static constexpr size_t OFF_XBF     = 0;                                      // T*D bf16
static constexpr size_t OFF_W1BF    = OFF_XBF  + (size_t)T_TOK * DIM_D * 2;   // E*D*H bf16
static constexpr size_t OFF_W2BF    = OFF_W1BF + (size_t)N_EXP * DIM_D * DIM_H * 2;
static constexpr size_t OFF_HBUF    = OFF_W2BF + (size_t)N_EXP * DIM_H * DIM_D * 2; // (2T)*H bf16
static constexpr size_t OFF_COUNTS  = OFF_HBUF + (size_t)T_TOK * 2 * DIM_H * 2;
static constexpr size_t OFF_OFFSETS = OFF_COUNTS + 64;
static constexpr size_t OFF_LISTS   = OFF_COUNTS + 256;                       // E*T ints
static constexpr size_t OFF_COEFS   = OFF_LISTS + (size_t)N_EXP * T_TOK * 4;  // E*T floats

__device__ __forceinline__ unsigned short f2bf(float f) {
    union { float f; unsigned int u; } c; c.f = f;
    unsigned int u = c.u + 0x7FFFu + ((c.u >> 16) & 1u);
    return (unsigned short)(u >> 16);
}

__device__ __forceinline__ float gelu_exact(float x) {
    return 0.5f * x * (1.0f + erff(x * 0.70710678118654752f));
}

// issue one per-lane 16B async copy global -> LDS (ASYNCcnt tracked)
__device__ __forceinline__ void async_cp16(unsigned lds_off, unsigned long long gaddr) {
    asm volatile("global_load_async_to_lds_b128 %0, %1, off"
                 :: "v"(lds_off), "v"(gaddr) : "memory");
}
__device__ __forceinline__ void async_wait0() {
    asm volatile("s_wait_asynccnt 0x0" ::: "memory");
}

// ---------------------------------------------------------------------------
__global__ void moe_zero_f32(float* __restrict__ p, size_t n) {
    size_t i = (size_t)blockIdx.x * blockDim.x + threadIdx.x;
    size_t stride = (size_t)gridDim.x * blockDim.x;
    for (; i < n; i += stride) p[i] = 0.0f;
}

__global__ void moe_cvt_bf16(const float* __restrict__ src,
                             unsigned short* __restrict__ dst, size_t n4) {
    size_t i = (size_t)blockIdx.x * blockDim.x + threadIdx.x;
    size_t stride = (size_t)gridDim.x * blockDim.x;
    for (; i < n4; i += stride) {
        float4 f = ((const float4*)src)[i];
        unsigned short o0 = f2bf(f.x), o1 = f2bf(f.y), o2 = f2bf(f.z), o3 = f2bf(f.w);
        unsigned long long pk = (unsigned long long)o0 | ((unsigned long long)o1 << 16) |
                                ((unsigned long long)o2 << 32) | ((unsigned long long)o3 << 48);
        ((unsigned long long*)dst)[i] = pk;
    }
}

// ---------------------------------------------------------------------------
// Gating: one wave per token; top-2 + softmax; compact per-expert lists.
__global__ void moe_gate(const float* __restrict__ x, const float* __restrict__ gw,
                         int* __restrict__ counts, int* __restrict__ lists,
                         float* __restrict__ coefs) {
    int lane = threadIdx.x & 31;
    int t = blockIdx.x * (blockDim.x >> 5) + (threadIdx.x >> 5);
    if (t >= T_TOK) return;

    float s[N_EXP];
#pragma unroll
    for (int e = 0; e < N_EXP; ++e) s[e] = 0.0f;

    const float* xr = x + (size_t)t * DIM_D;
    for (int d = lane; d < DIM_D; d += 32) {
        float xv = xr[d];
        const float* g = gw + (size_t)d * N_EXP;
#pragma unroll
        for (int e = 0; e < N_EXP; ++e) s[e] += xv * g[e];
    }
#pragma unroll
    for (int off = 16; off > 0; off >>= 1)
#pragma unroll
        for (int e = 0; e < N_EXP; ++e) s[e] += __shfl_xor(s[e], off, 32);

    if (lane == 0) {
        int i0 = 0;
#pragma unroll
        for (int e = 1; e < N_EXP; ++e) if (s[e] > s[i0]) i0 = e;
        int i1 = (i0 == 0) ? 1 : 0;
#pragma unroll
        for (int e = 0; e < N_EXP; ++e)
            if (e != i0 && s[e] > s[i1]) i1 = e;
        float e1 = __expf(s[i1] - s[i0]);
        float z  = 1.0f + e1;
        float c0 = 1.0f / z, c1 = e1 / z;

        int s0 = atomicAdd(&counts[i0], 1);
        lists[i0 * T_TOK + s0] = t;
        coefs[i0 * T_TOK + s0] = c0;
        int s1 = atomicAdd(&counts[i1], 1);
        lists[i1 * T_TOK + s1] = t;
        coefs[i1 * T_TOK + s1] = c1;
    }
}

__global__ void moe_scan(const int* __restrict__ counts, int* __restrict__ offsets) {
    if (threadIdx.x == 0 && blockIdx.x == 0) {
        int acc = 0;
        offsets[0] = 0;
        for (int e = 0; e < N_EXP; ++e) { acc += counts[e]; offsets[e + 1] = acc; }
    }
}

// ---------------------------------------------------------------------------
// Shared GEMM tile body pieces -----------------------------------------------
// B loader: stage [Kc=64][256] weight chunk into LDS transposed as lds[n][k].
// Each thread owns a column pair (n2, n2+1) and 32 K rows; b32 global reads
// cover both columns, fully coalesced across lanes.
__device__ __forceinline__ void load_B_tile(const unsigned short* __restrict__ g,
                                            int ldN, int tid,
                                            unsigned short (*ldsB)[64]) {
    int n2 = (tid & 127) * 2;
    int kq = (tid >> 7) * 32;
    const unsigned short* gp = g + (size_t)kq * ldN + n2;
#pragma unroll
    for (int c = 0; c < 4; ++c) {
        PackU lo, hi;
#pragma unroll
        for (int j = 0; j < 8; ++j) {
            unsigned int w = *(const unsigned int*)(gp + (size_t)(c * 8 + j) * ldN);
            lo.s[j] = (unsigned short)(w & 0xffffu);
            hi.s[j] = (unsigned short)(w >> 16);
        }
        *(u32x4*)&ldsB[n2][kq + c * 8]     = lo.q;
        *(u32x4*)&ldsB[n2 + 1][kq + c * 8] = hi.q;
    }
}

// ---------------------------------------------------------------------------
// GEMM1: hbuf[pair, h] = gelu( x[tok] @ w1[e] + b1[e] )   (bf16 out)
__global__ __launch_bounds__(256)
void moe_ffn1(const unsigned short* __restrict__ xbf,
              const unsigned short* __restrict__ w1bf,
              const float* __restrict__ b1,
              const int* __restrict__ counts, const int* __restrict__ offsets,
              const int* __restrict__ lists, unsigned short* __restrict__ hbuf) {
    const int e  = blockIdx.z;
    const int mt = blockIdx.x;
    const int nt = blockIdx.y;
    const int cnt = counts[e];
    if (mt * 128 >= cnt) return;
    const int offe = offsets[e];
    const int* list = lists + e * T_TOK;

    __shared__ int stok[128];
    __shared__ __align__(16) unsigned short ldsA[128][64];  // [m][k] 16 KB
    __shared__ __align__(16) unsigned short ldsB[256][64];  // [n][k] 32 KB

    const int tid = threadIdx.x;
    if (tid < 128) {
        int r = mt * 128 + tid;
        stok[tid] = list[(r < cnt) ? r : (cnt - 1)];
    }
    __syncthreads();

    const unsigned short* Wp = w1bf + (size_t)e * DIM_D * DIM_H;
    const int n0 = nt * 256;
    const int lane  = tid & 31;
    const int waveM = (tid >> 5) & 1;
    const int waveN = (tid >> 6);

    v8f acc[4][4];
#pragma unroll
    for (int i = 0; i < 4; ++i)
#pragma unroll
        for (int j = 0; j < 4; ++j) acc[i][j] = (v8f)(0.0f);

    const int rA    = tid >> 1;
    const int halfA = tid & 1;
    const unsigned ldsA_off =
        (unsigned)(size_t)(lds_us*)&ldsA[rA][halfA * 32];
    const unsigned long long gA_row =
        (unsigned long long)(size_t)(xbf + (size_t)stok[rA] * DIM_D + halfA * 32);

    for (int k0 = 0; k0 < DIM_D; k0 += 64) {
        __syncthreads();
        // A tile: async global -> LDS, 4 x 16B per thread (ASYNCcnt path)
#pragma unroll
        for (int c = 0; c < 4; ++c)
            async_cp16(ldsA_off + c * 16, gA_row + (size_t)k0 * 2 + c * 16);
        // B tile: VGPR-staged transpose
        load_B_tile(Wp + (size_t)k0 * DIM_H + n0, DIM_H, tid, ldsB);
        async_wait0();
        __syncthreads();

        const int khA = (lane < 16) ? 0 : 8;
        const int khB = (lane < 16) ? 0 : 8;
#pragma unroll
        for (int kk = 0; kk < 64; kk += 32) {
            Frag bf[4];
#pragma unroll
            for (int nn = 0; nn < 4; ++nn) {
                int n = waveN * 64 + nn * 16 + (lane & 15);
                bf[nn].q[0] = *(const u32x4*)&ldsB[n][kk + khB];
                bf[nn].q[1] = *(const u32x4*)&ldsB[n][kk + khB + 16];
            }
#pragma unroll
            for (int mm = 0; mm < 4; ++mm) {
                Frag af;
                int m = waveM * 64 + mm * 16 + (lane & 15);
                af.q[0] = *(const u32x4*)&ldsA[m][kk + khA];
                af.q[1] = *(const u32x4*)&ldsA[m][kk + khA + 16];
#pragma unroll
                for (int nn = 0; nn < 4; ++nn)
                    acc[mm][nn] = __builtin_amdgcn_wmma_f32_16x16x32_bf16(
                        false, af.v, false, bf[nn].v, (short)0, acc[mm][nn], false, false);
            }
        }
    }

    // epilogue: + b1, exact GeLU, bf16 store
#pragma unroll
    for (int mm = 0; mm < 4; ++mm)
#pragma unroll
        for (int nn = 0; nn < 4; ++nn)
#pragma unroll
            for (int v = 0; v < 8; ++v) {
                int ml = waveM * 64 + mm * 16 + ((lane >> 4) << 3) + v;
                int r  = mt * 128 + ml;
                if (r < cnt) {
                    int col = n0 + waveN * 64 + nn * 16 + (lane & 15);
                    float val = acc[mm][nn][v] + b1[e * DIM_H + col];
                    hbuf[(size_t)(offe + r) * DIM_H + col] = f2bf(gelu_exact(val));
                }
            }
}

// ---------------------------------------------------------------------------
// GEMM2: y[tok] += coef * ( h @ w2[e] + b2[e] )  via global f32 atomics.
__global__ __launch_bounds__(256)
void moe_ffn2(const unsigned short* __restrict__ hbuf,
              const unsigned short* __restrict__ w2bf,
              const float* __restrict__ b2,
              const int* __restrict__ counts, const int* __restrict__ offsets,
              const int* __restrict__ lists, const float* __restrict__ coefs,
              float* __restrict__ y) {
    const int e  = blockIdx.z;
    const int mt = blockIdx.x;
    const int nt = blockIdx.y;
    const int cnt = counts[e];
    if (mt * 128 >= cnt) return;
    const int offe = offsets[e];

    __shared__ int   stok[128];
    __shared__ float scoef[128];
    __shared__ __align__(16) unsigned short ldsA[128][64];
    __shared__ __align__(16) unsigned short ldsB[256][64];

    const int tid = threadIdx.x;
    if (tid < 128) {
        int r = mt * 128 + tid;
        int rc = (r < cnt) ? r : (cnt - 1);
        stok[tid]  = lists[e * T_TOK + rc];
        scoef[tid] = (r < cnt) ? coefs[e * T_TOK + rc] : 0.0f;
    }
    __syncthreads();

    const unsigned short* Wp = w2bf + (size_t)e * DIM_H * DIM_D;
    const int n0 = nt * 256;
    const int lane  = tid & 31;
    const int waveM = (tid >> 5) & 1;
    const int waveN = (tid >> 6);

    v8f acc[4][4];
#pragma unroll
    for (int i = 0; i < 4; ++i)
#pragma unroll
        for (int j = 0; j < 4; ++j) acc[i][j] = (v8f)(0.0f);

    const int rA    = tid >> 1;
    const int halfA = tid & 1;
    int rr = mt * 128 + rA; if (rr >= cnt) rr = cnt - 1;
    const unsigned ldsA_off =
        (unsigned)(size_t)(lds_us*)&ldsA[rA][halfA * 32];
    const unsigned long long gA_row =
        (unsigned long long)(size_t)(hbuf + (size_t)(offe + rr) * DIM_H + halfA * 32);

    for (int k0 = 0; k0 < DIM_H; k0 += 64) {
        __syncthreads();
#pragma unroll
        for (int c = 0; c < 4; ++c)
            async_cp16(ldsA_off + c * 16, gA_row + (size_t)k0 * 2 + c * 16);
        load_B_tile(Wp + (size_t)k0 * DIM_D + n0, DIM_D, tid, ldsB);
        async_wait0();
        __syncthreads();

        const int khA = (lane < 16) ? 0 : 8;
        const int khB = (lane < 16) ? 0 : 8;
#pragma unroll
        for (int kk = 0; kk < 64; kk += 32) {
            Frag bf[4];
#pragma unroll
            for (int nn = 0; nn < 4; ++nn) {
                int n = waveN * 64 + nn * 16 + (lane & 15);
                bf[nn].q[0] = *(const u32x4*)&ldsB[n][kk + khB];
                bf[nn].q[1] = *(const u32x4*)&ldsB[n][kk + khB + 16];
            }
#pragma unroll
            for (int mm = 0; mm < 4; ++mm) {
                Frag af;
                int m = waveM * 64 + mm * 16 + (lane & 15);
                af.q[0] = *(const u32x4*)&ldsA[m][kk + khA];
                af.q[1] = *(const u32x4*)&ldsA[m][kk + khA + 16];
#pragma unroll
                for (int nn = 0; nn < 4; ++nn)
                    acc[mm][nn] = __builtin_amdgcn_wmma_f32_16x16x32_bf16(
                        false, af.v, false, bf[nn].v, (short)0, acc[mm][nn], false, false);
            }
        }
    }

    // epilogue: coef * (acc + b2) atomically accumulated into y[token]
#pragma unroll
    for (int mm = 0; mm < 4; ++mm)
#pragma unroll
        for (int nn = 0; nn < 4; ++nn)
#pragma unroll
            for (int v = 0; v < 8; ++v) {
                int ml = waveM * 64 + mm * 16 + ((lane >> 4) << 3) + v;
                int r  = mt * 128 + ml;
                if (r < cnt) {
                    int col = n0 + waveN * 64 + nn * 16 + (lane & 15);
                    float val = scoef[ml] * (acc[mm][nn][v] + b2[e * DIM_D + col]);
                    atomicAdd(&y[(size_t)stok[ml] * DIM_D + col], val);
                }
            }
}

// ---------------------------------------------------------------------------
extern "C" void kernel_launch(void* const* d_in, const int* in_sizes, int n_in,
                              void* d_out, int out_size, void* d_ws, size_t ws_size,
                              hipStream_t stream) {
    const float* x      = (const float*)d_in[0];
    const float* gate_w = (const float*)d_in[1];
    const float* w1     = (const float*)d_in[2];
    const float* b1     = (const float*)d_in[3];
    const float* w2     = (const float*)d_in[4];
    const float* b2     = (const float*)d_in[5];
    float* y = (float*)d_out;

    char* ws = (char*)d_ws;
    unsigned short* xbf  = (unsigned short*)(ws + OFF_XBF);
    unsigned short* w1bf = (unsigned short*)(ws + OFF_W1BF);
    unsigned short* w2bf = (unsigned short*)(ws + OFF_W2BF);
    unsigned short* hbuf = (unsigned short*)(ws + OFF_HBUF);
    int*   counts  = (int*)(ws + OFF_COUNTS);
    int*   offsets = (int*)(ws + OFF_OFFSETS);
    int*   lists   = (int*)(ws + OFF_LISTS);
    float* coefs   = (float*)(ws + OFF_COEFS);

    moe_zero_f32<<<2048, 256, 0, stream>>>(y, (size_t)T_TOK * DIM_D);
    moe_zero_f32<<<1, 64, 0, stream>>>((float*)counts, 16);

    moe_cvt_bf16<<<2048, 256, 0, stream>>>(x,  xbf,  (size_t)T_TOK * DIM_D / 4);
    moe_cvt_bf16<<<4096, 256, 0, stream>>>(w1, w1bf, (size_t)N_EXP * DIM_D * DIM_H / 4);
    moe_cvt_bf16<<<4096, 256, 0, stream>>>(w2, w2bf, (size_t)N_EXP * DIM_H * DIM_D / 4);

    moe_gate<<<T_TOK / 8, 256, 0, stream>>>(x, gate_w, counts, lists, coefs);
    moe_scan<<<1, 32, 0, stream>>>(counts, offsets);

    dim3 gA(T_TOK / 128, DIM_H / 256, N_EXP);
    moe_ffn1<<<gA, 256, 0, stream>>>(xbf, w1bf, b1, counts, offsets, lists, hbuf);
    dim3 gB(T_TOK / 128, DIM_D / 256, N_EXP);
    moe_ffn2<<<gB, 256, 0, stream>>>(hbuf, w2bf, b2, counts, offsets, lists, coefs, y);
}
